// CausalStructureEnhancedGAT_80324478370389
// MI455X (gfx1250) — matrix operand order repaired and checked by reading, not compiled, not run on hardware
//
#include <hip/hip_runtime.h>
#include <hip/hip_fp16.h>

#define B_ 8
#define N_ 2048
#define DIN_ 128
#define DOUT_ 64
#define H_ 4
#define NEG_SLOPE_ 0.2f

typedef __attribute__((ext_vector_type(16))) _Float16 v16h;
typedef __attribute__((ext_vector_type(8)))  float    v8f;
typedef __attribute__((ext_vector_type(4)))  int      i32x4;

#define AS1 __attribute__((address_space(1)))
#define AS3 __attribute__((address_space(3)))

union F16x16 { uint4 u[2]; v16h v; };

#if __has_builtin(__builtin_amdgcn_global_load_async_to_lds_b128)
#define HAS_ASYNC_LDS 1
#else
#define HAS_ASYNC_LDS 0
#endif

// ---------------------------------------------------------------------------
// Kernel 1: m[n] = mean_j cs[n, j]
// ---------------------------------------------------------------------------
__global__ void k_rowmean(const float* __restrict__ cs, float* __restrict__ m) {
    __shared__ float red[256];
    const int n = blockIdx.x;
    const int t = threadIdx.x;
    float s = 0.f;
    for (int j = t; j < N_; j += 256) s += cs[n * N_ + j];
    red[t] = s;
    __syncthreads();
    for (int off = 128; off > 0; off >>= 1) {
        if (t < off) red[t] += red[t + off];
        __syncthreads();
    }
    if (t == 0) m[n] = red[0] * (1.0f / (float)N_);
}

// ---------------------------------------------------------------------------
// Kernel 2: x2 = x + (x . ct_w^T + ct_b) * m[n]   (one row per block, fp32)
// ---------------------------------------------------------------------------
__global__ void __launch_bounds__(128) k_causal_transform(
    const float* __restrict__ x, const float* __restrict__ ct_w,
    const float* __restrict__ ct_b, const float* __restrict__ m,
    float* __restrict__ x2) {
    __shared__ float xs[DIN_];
    const int row = blockIdx.x;      // b*N + n
    const int n   = row % N_;
    const int e   = threadIdx.x;     // output feature index 0..127
    xs[e] = x[(size_t)row * DIN_ + e];
    __syncthreads();
    float acc = ct_b[e];
    const float* w = ct_w + (size_t)e * DIN_;   // ct_w[e, d]
    #pragma unroll 8
    for (int d = 0; d < DIN_; ++d) acc += xs[d] * w[d];
    x2[(size_t)row * DIN_ + e] = xs[e] + acc * m[n];
}

// ---------------------------------------------------------------------------
// Kernel 3: xt[b,h,n,:] = x2[b,n,:] . W[h]; s_i/s_j row dots.  xt stored f16.
// (s_i/s_j computed in fp32 before the f16 downcast -> full score fidelity)
// ---------------------------------------------------------------------------
__global__ void __launch_bounds__(64) k_project(
    const float* __restrict__ x2, const float* __restrict__ W,
    const float* __restrict__ att, _Float16* __restrict__ xt_h,
    float* __restrict__ s_i, float* __restrict__ s_j) {
    __shared__ float xs[DIN_];
    __shared__ float r0[DOUT_], r1[DOUT_];
    const int n  = blockIdx.x;
    const int bh = blockIdx.y;
    const int b  = bh / H_, h = bh % H_;
    const int o  = threadIdx.x;      // 0..63
    xs[o]      = x2[((size_t)b * N_ + n) * DIN_ + o];
    xs[o + 64] = x2[((size_t)b * N_ + n) * DIN_ + o + 64];
    __syncthreads();
    float acc = 0.f;
    const float* w = W + ((size_t)h * DIN_) * DOUT_ + o;   // W[h, d, o]
    #pragma unroll 8
    for (int d = 0; d < DIN_; ++d) acc += xs[d] * w[(size_t)d * DOUT_];
    xt_h[((size_t)bh * N_ + n) * DOUT_ + o] = (_Float16)acc;
    r0[o] = acc * att[h * (2 * DOUT_) + o];          // a_src
    r1[o] = acc * att[h * (2 * DOUT_) + DOUT_ + o];  // a_dst
    __syncthreads();
    if (o < 32) { r0[o] += r0[o + 32]; r1[o] += r1[o + 32]; }
    __syncthreads();
    if (o == 0) {
        float a0 = 0.f, a1 = 0.f;
        #pragma unroll 8
        for (int k = 0; k < 32; ++k) { a0 += r0[k]; a1 += r1[k]; }
        s_i[(size_t)bh * N_ + n] = a0;
        s_j[(size_t)bh * N_ + n] = a1;
    }
}

// ---------------------------------------------------------------------------
// Kernel 4: fused masked-softmax attention (flash-style) with
//   - async global->LDS staging of the V tile (ASYNCcnt path)
//   - WMMA f16 P.V accumulation (f32 accumulators)
//   - WMMA f16 sigmoid-gate GEMM epilogue, fully in registers
//   - fused (0,2,1,3) transpose on store
// One block = 16 query rows of one (b,h); 4 waves, each owns 16 output cols.
// ---------------------------------------------------------------------------
__global__ void __launch_bounds__(128) k_attention(
    const float* __restrict__ cs, const float* __restrict__ bias_p,
    const _Float16* __restrict__ xt_h, const float* __restrict__ s_i,
    const float* __restrict__ s_j, const float* __restrict__ cg_w,
    const float* __restrict__ cg_b, float* __restrict__ out) {

    __shared__ __align__(16) float    s_tile[16 * 32];
    __shared__ __align__(16) _Float16 p_tile[16 * 32];       // P, [i][k] row-major
    __shared__ __align__(16) _Float16 v_tile[32 * DOUT_];    // V, [j][o] row-major
    __shared__ __align__(16) _Float16 o_half[16 * DOUT_];    // normalized out, f16
    __shared__ __align__(16) _Float16 cw_half[DOUT_ * DOUT_];// cg_w as f16 (B layout)
    __shared__ float m_row[16], l_row[16], scale_row[16], si_row[16];

    const int t    = threadIdx.x;
    const int lane = t & 31;
    const int wave = t >> 5;                  // 0..3 -> 16-wide o-chunk
    const int i0   = blockIdx.x * 16;
    const int bh   = blockIdx.y;
    const int h    = bh % H_;
    const int b    = bh / H_;
    const float bias = bias_p[0];
    const float NEG_INF = -__builtin_inff();

    const int mrow = lane & 15;
    const int kg   = (lane >= 16) ? 1 : 0;    // K half-group select for frags
    const int rbase = kg ? 8 : 0;             // C layout: M = r + 8*(lane>=16)
    const int ncol = wave * 16 + mrow;        // this lane's output column

    // stage cg_w as f16; its row-major layout IS the WMMA B(k,n) layout
    #pragma unroll
    for (int q = 0; q < 32; ++q)
        cw_half[t * 32 + q] = (_Float16)cg_w[t * 32 + q];

    if (t < 16) {
        m_row[t]  = NEG_INF;
        l_row[t]  = 0.f;
        si_row[t] = s_i[(size_t)bh * N_ + i0 + t];
    }
    __syncthreads();

    v8f acc = {};

    for (int j0 = 0; j0 < N_; j0 += 32) {
        // -- V tile (32 x 64 f16 = 4KB, contiguous in memory): async -> LDS --
        const _Float16* gsrc = xt_h + ((size_t)bh * N_ + j0) * DOUT_ + t * 16;
#if HAS_ASYNC_LDS
        __builtin_amdgcn_global_load_async_to_lds_b128(
            (AS1 i32x4*)(AS1 void*)gsrc,
            (AS3 i32x4*)(AS3 void*)&v_tile[t * 16], 0, 0);
        __builtin_amdgcn_global_load_async_to_lds_b128(
            (AS1 i32x4*)(AS1 void*)gsrc,
            (AS3 i32x4*)(AS3 void*)&v_tile[t * 16], 16, 0);
#else
        {
            const uint4* src = reinterpret_cast<const uint4*>(gsrc);
            uint4* dst = reinterpret_cast<uint4*>(&v_tile[t * 16]);
            dst[0] = src[0];
            dst[1] = src[1];
        }
#endif
        // -- score tile (16 x 32), 4 elements per thread (overlaps async) --
        #pragma unroll
        for (int r = 0; r < 4; ++r) {
            const int idx = t + 128 * r;
            const int il  = idx >> 5;
            const int jl  = idx & 31;
            const float csv = cs[(size_t)(i0 + il) * N_ + j0 + jl];
            const float raw = si_row[il] + s_j[(size_t)bh * N_ + j0 + jl];
            const float lr  = raw >= 0.f ? raw : NEG_SLOPE_ * raw;
            const float sc  = lr + bias * csv;
            s_tile[idx] = (csv == 0.f) ? NEG_INF : sc;
        }
#if HAS_ASYNC_LDS
#if __has_builtin(__builtin_amdgcn_s_wait_asynccnt)
        __builtin_amdgcn_s_wait_asynccnt(0);
#else
        asm volatile("s_wait_asynccnt 0" ::: "memory");
#endif
#endif
        __syncthreads();

        // -- online-softmax row update (threads 0..15, one row each) --
        if (t < 16) {
            const float mold = m_row[t];
            float rmax = NEG_INF;
            #pragma unroll 4
            for (int j = 0; j < 32; ++j) rmax = fmaxf(rmax, s_tile[t * 32 + j]);
            const float mnew = fmaxf(mold, rmax);
            float scale, psum = 0.f;
            if (mnew == NEG_INF) {            // fully-masked so far
                scale = 1.f;
                for (int j = 0; j < 32; ++j) p_tile[t * 32 + j] = (_Float16)0.f;
            } else {
                scale = (mold == NEG_INF) ? 0.f : __expf(mold - mnew);
                for (int j = 0; j < 32; ++j) {
                    const float sv = s_tile[t * 32 + j];
                    const float p  = (sv == NEG_INF) ? 0.f : __expf(sv - mnew);
                    psum += p;
                    p_tile[t * 32 + j] = (_Float16)p;
                }
            }
            l_row[t]     = l_row[t] * scale + psum;
            m_row[t]     = mnew;
            scale_row[t] = scale;
        }
        __syncthreads();

        // -- rescale accumulator, build fragments, WMMA (EXEC all-ones) --
        {
            #pragma unroll
            for (int r = 0; r < 8; ++r) acc[r] *= scale_row[r + rbase];

            // A fragment = two contiguous 16B runs of this lane's P row:
            //   k in [8*kg, 8*kg+8) and [16+8*kg, 16+8*kg+8)
            F16x16 au;
            au.u[0] = *reinterpret_cast<const uint4*>(&p_tile[mrow * 32 + kg * 8]);
            au.u[1] = *reinterpret_cast<const uint4*>(&p_tile[mrow * 32 + kg * 8 + 16]);

            // B fragment: B(k,n); k = 8*kg + e (e<8), 16 + 8*kg + (e-8) (e>=8), n = ncol  (column gather)
            v16h bfrag;
            const int kb = kg * 8;
            #pragma unroll
            for (int e = 0; e < 16; ++e)
                bfrag[e] = v_tile[(kb + (e < 8 ? e : e + 8)) * DOUT_ + ncol];

            acc = __builtin_amdgcn_wmma_f32_16x16x32_f16(
                false, au.v, false, bfrag, (short)0, acc, false, false);
        }
        __syncthreads();   // protect p_tile / v_tile for next iteration
    }

    // -- normalize in registers; stage f16 copy as gate-GEMM A matrix --
    v8f oreg;
    #pragma unroll
    for (int r = 0; r < 8; ++r) {
        const int row = r + rbase;
        oreg[r] = acc[r] / l_row[row];
        o_half[row * DOUT_ + ncol] = (_Float16)oreg[r];
    }
    __syncthreads();

    // -- gate GEMM: g[i][p] = sum_o out[i][o] * cg_w[p][o]; two WMMA k-steps --
    v8f gacc = {};
    #pragma unroll
    for (int s = 0; s < 2; ++s) {
        F16x16 au;
        const int kbase = s * 32;
        au.u[0] = *reinterpret_cast<const uint4*>(&o_half[mrow * DOUT_ + kbase + kg * 16]);
        au.u[1] = *reinterpret_cast<const uint4*>(&o_half[mrow * DOUT_ + kbase + kg * 16 + 8]);
        F16x16 bu;   // 16 contiguous halves of cg_w row `ncol`
        bu.u[0] = *reinterpret_cast<const uint4*>(&cw_half[ncol * DOUT_ + kbase + kg * 16]);
        bu.u[1] = *reinterpret_cast<const uint4*>(&cw_half[ncol * DOUT_ + kbase + kg * 16 + 8]);
        gacc = __builtin_amdgcn_wmma_f32_16x16x32_f16(
            false, au.v, false, bu.v, (short)0, gacc, false, false);
    }

    // -- sigmoid gate, apply, fused (0,2,1,3) transpose store (registers only) --
    const float cgb = cg_b[ncol];
    #pragma unroll
    for (int r = 0; r < 8; ++r) {
        const int row = r + rbase;
        const float g   = gacc[r] + cgb;
        const float sig = 1.f / (1.f + __expf(-g));
        out[(((size_t)b * N_ + (i0 + row)) * H_ + h) * DOUT_ + ncol] = oreg[r] * sig;
    }
}

// ---------------------------------------------------------------------------
// Host launcher
// ---------------------------------------------------------------------------
extern "C" void kernel_launch(void* const* d_in, const int* in_sizes, int n_in,
                              void* d_out, int out_size, void* d_ws, size_t ws_size,
                              hipStream_t stream) {
    (void)in_sizes; (void)n_in; (void)out_size; (void)ws_size;
    const float* x     = (const float*)d_in[0];
    const float* cs    = (const float*)d_in[1];
    const float* W     = (const float*)d_in[2];
    const float* att   = (const float*)d_in[3];
    const float* bias  = (const float*)d_in[4];
    const float* ct_w  = (const float*)d_in[5];
    const float* ct_b  = (const float*)d_in[6];
    const float* cg_w  = (const float*)d_in[7];
    const float* cg_b  = (const float*)d_in[8];
    float* out = (float*)d_out;

    // workspace layout (all offsets 16B-aligned)
    char* ws = (char*)d_ws;
    float*    m_row = (float*)ws;                                   //  8 KB
    float*    x2    = (float*)(ws + 8192);                          //  8 MB
    float*    s_i   = (float*)(ws + 8192 + (size_t)8 * 1024 * 1024);           // 256 KB
    float*    s_j   = (float*)(ws + 8192 + (size_t)8 * 1024 * 1024 + 262144);  // 256 KB
    _Float16* xt_h  = (_Float16*)(ws + 8192 + (size_t)8 * 1024 * 1024 + 524288); // 8 MB

    k_rowmean<<<N_, 256, 0, stream>>>(cs, m_row);
    k_causal_transform<<<B_ * N_, 128, 0, stream>>>(x, ct_w, ct_b, m_row, x2);
    k_project<<<dim3(N_, B_ * H_), 64, 0, stream>>>(x2, W, att, xt_h, s_i, s_j);
    k_attention<<<dim3(N_ / 16, B_ * H_), 128, 0, stream>>>(
        cs, bias, xt_h, s_i, s_j, cg_w, cg_b, out);
}
